// Generator_T2I_56599079027112
// MI455X (gfx1250) — compile-verified, matrix operand order checked by reading, not run on hardware
//
#include <hip/hip_runtime.h>
#include <hip/hip_bf16.h>

// ---------------------------------------------------------------------------
// MI455X (gfx1250) implementation.
//
// Precision strategy: all GEMMs run on v_wmma_f32_16x16x32_bf16 (bf16 in,
// fp32 accumulate). GRU master hidden state is kept in fp32 registers; only
// the WMMA operand mirror of h is bf16.
//
// gfx1250-specific paths used:
//   * v_wmma_f32_16x16x32_bf16 for every GEMM
//   * global_load_async_to_lds_b128 (+ s_wait_asynccnt) to stage B-tiles in LDS
//   * global_prefetch_b8 hints on the streaming A operand
//   * 256 KB of the 320 KB WGP LDS holds the whole GRU step state
//
// Workspace layout (bytes; total ~474 MB):
//   Wcat  bf16 [1536][256]          @ 0          (786,432)
//   Wlin  bf16 [128][256]           @ 786,432    (65,536)
//   W1T   bf16 [25088][256]         @ 851,968    (12,845,056)
//   Xbf   bf16 [131072][256]        @ 13,697,024 (67,108,864)
//   XP    f32  [131072][768]        @ 80,805,888 (402,653,184)
//   HT    f32  [128][256]           @ 483,459,072 (131,072)
//   FEAT  bf16 [128][256]           @ 483,590,144 (65,536)
//   Y     f32  [128][25088]         @ 483,655,680 (12,845,056)
// ---------------------------------------------------------------------------

typedef __bf16 bf16_t;
typedef __attribute__((ext_vector_type(16))) __bf16 v16bf;
typedef __attribute__((ext_vector_type(8)))  __bf16 v8bf;
typedef __attribute__((ext_vector_type(8)))  float  v8f;

__device__ __forceinline__ v8f vzero8() {
  v8f z;
#pragma unroll
  for (int i = 0; i < 8; ++i) z[i] = 0.f;
  return z;
}

// Async DMA of one 16-byte chunk global -> LDS (ASYNCcnt-tracked).
// lds_addr = wave-relative LDS byte offset = low 32 bits of the generic
// address of a __shared__ object (generic LDS addr = {SHARED_BASE, off[31:0]}).
__device__ __forceinline__ void async_g2l_b128(unsigned lds_addr, const void* gaddr) {
  asm volatile("global_load_async_to_lds_b128 %0, %1, off"
               :: "v"(lds_addr), "v"(gaddr)
               : "memory");
}
__device__ __forceinline__ void wait_async0() {
  asm volatile("s_wait_asynccnt 0x0" ::: "memory");
}
__device__ __forceinline__ unsigned lds_off(const void* p) {
  return (unsigned)(size_t)p;       // truncate generic addr -> LDS byte offset
}

// Load a 16x32 bf16 WMMA A/B fragment from a row-major [*, ld] matrix whose
// tile origin is `base` (works for global and LDS pointers).
// Per CDNA5 ISA 16-bit operand layout:
//  lane L: row m = L&15, half = L>>4; elems 0..7 -> K = half*8 + i,
//  elems 8..15 -> K = 16 + half*8 + i.  Two 16-byte loads per lane.
__device__ __forceinline__ v16bf frag_bf16(const bf16_t* base, int ld, int lane) {
  int m = lane & 15, half = lane >> 4;
  const bf16_t* p = base + (size_t)m * ld + half * 8;
  v8bf lo = *reinterpret_cast<const v8bf*>(p);
  v8bf hi = *reinterpret_cast<const v8bf*>(p + 16);
  return __builtin_shufflevector(lo, hi, 0, 1, 2, 3, 4, 5, 6, 7,
                                 8, 9, 10, 11, 12, 13, 14, 15);
}

// Same fragment but converting from an fp32 source on the fly.
__device__ __forceinline__ v16bf frag_f32(const float* base, int ld, int lane) {
  int m = lane & 15, half = lane >> 4;
  const float* p = base + (size_t)m * ld + half * 8;
  v16bf a;
#pragma unroll
  for (int i = 0; i < 8; ++i) {
    a[i]     = (bf16_t)p[i];
    a[i + 8] = (bf16_t)p[16 + i];
  }
  return a;
}

__device__ __forceinline__ v8f wmma_bf(v16bf a, v16bf b, v8f c) {
  // v_wmma_f32_16x16x32_bf16  (neg_a, A, neg_b, B, c_mod, C, reuse_a, reuse_b)
  return __builtin_amdgcn_wmma_f32_16x16x32_bf16(false, a, false, b,
                                                 (short)0, c, false, false);
}

__device__ __forceinline__ float sigmoidf_(float x) {
  return 1.f / (1.f + __expf(-x));
}

// ------------------------------ pack kernels -------------------------------

__global__ void k_pack_wcat(const float* __restrict__ w_ih,
                            const float* __restrict__ w_hh,
                            bf16_t* __restrict__ Wcat) {
  int r = blockIdx.x, c = threadIdx.x;            // grid 1536 x 256
  float v = (r < 768) ? w_ih[r * 256 + c] : w_hh[(r - 768) * 256 + c];
  Wcat[r * 256 + c] = (bf16_t)v;
}

__global__ void k_pack_wlin(const float* __restrict__ w_lin,
                            bf16_t* __restrict__ Wlin) {
  int i = blockIdx.x * 256 + threadIdx.x;         // grid 128 -> 32768
  Wlin[i] = (bf16_t)w_lin[i];
}

__global__ void k_pack_w1t(const float* __restrict__ W1,
                           bf16_t* __restrict__ W1T) {
  size_t idx = (size_t)blockIdx.x * 256 + threadIdx.x;  // grid 25088 -> 6,422,528
  int i = (int)(idx / 25088);
  int n = (int)(idx % 25088);
  W1T[(size_t)n * 256 + i] = (bf16_t)W1[idx];     // coalesced read, scatter write
}

__global__ void k_cvt_x(const float* __restrict__ x, bf16_t* __restrict__ Xbf) {
  size_t idx = (size_t)blockIdx.x * 256 + threadIdx.x;  // grid 131072
  Xbf[idx] = (bf16_t)x[idx];
}

// --------------------------- x_proj GEMM (WMMA) ----------------------------
// XP[M=131072, N=768] = Xbf @ Wihᵀ + b_ih ; tile 128x96 per WG, 8 waves.
// B-tile (96x256 bf16 = 48 KB) staged once into LDS via async DMA.
__global__ __launch_bounds__(256) void k_xproj(const bf16_t* __restrict__ X,
                                               const bf16_t* __restrict__ Wih,
                                               const float* __restrict__ b_ih,
                                               float* __restrict__ xp) {
  __shared__ bf16_t Bs[96 * 256];                  // 48 KB
  int lane = threadIdx.x & 31, wave = threadIdx.x >> 5;
  int wm = wave >> 1, wn = wave & 1;               // 4 x 2 wave grid
  int r0 = blockIdx.x * 128 + wm * 32;
  int c0 = blockIdx.y * 96;

  // async-stage the whole B block: 3072 16-byte chunks, 12 per thread
  {
    unsigned lb = lds_off(&Bs[0]);
    const char* gsrc = (const char*)(Wih + (size_t)c0 * 256);
#pragma unroll
    for (int i = 0; i < 12; ++i) {
      int chunk = threadIdx.x + i * 256;
      async_g2l_b128(lb + chunk * 16, gsrc + (size_t)chunk * 16);
    }
    wait_async0();
  }
  __syncthreads();

  v8f acc[2][3];
#pragma unroll
  for (int i = 0; i < 2; ++i)
#pragma unroll
    for (int j = 0; j < 3; ++j) acc[i][j] = vzero8();

#pragma unroll
  for (int kk = 0; kk < 256; kk += 32) {
    // hint the streaming A operand into near cache (global_prefetch_b8)
    if (kk + 32 < 256)
      __builtin_prefetch(X + (size_t)(r0 + (lane & 15)) * 256 + kk + 32, 0, 3);
    v16bf a0 = frag_bf16(X + (size_t)r0 * 256 + kk, 256, lane);
    v16bf a1 = frag_bf16(X + (size_t)(r0 + 16) * 256 + kk, 256, lane);
#pragma unroll
    for (int j = 0; j < 3; ++j) {
      v16bf bj = frag_bf16(&Bs[(size_t)(wn * 48 + j * 16) * 256 + kk], 256, lane);
      acc[0][j] = wmma_bf(a0, bj, acc[0][j]);
      acc[1][j] = wmma_bf(a1, bj, acc[1][j]);
    }
  }

  int nlo = lane & 15, mhi = (lane >> 4) * 8;
#pragma unroll
  for (int i = 0; i < 2; ++i)
#pragma unroll
    for (int j = 0; j < 3; ++j) {
      int c = c0 + wn * 48 + j * 16 + nlo;
      float bias = b_ih[c];
#pragma unroll
      for (int e = 0; e < 8; ++e) {
        int r = r0 + i * 16 + mhi + e;
        xp[(size_t)r * 768 + c] = acc[i][j][e] + bias;
      }
    }
}

// ------------------------------ recurrent GRU ------------------------------
// One 1024-thread workgroup (32 waves) on one WGP.  Per step:
//   hp[128,768] = bf16(h) @ Whhᵀ   (384 WMMA tiles, 12 per wave)
//   gate math in fp32 registers (master h lives in 32 regs/thread)
// LDS: h mirror bf16 64 KB + hp bf16 192 KB = 256 KB of the 320 KB WGP LDS.
// Whh (384 KB bf16) stays in L2/WGP$ -- it cannot fit next to the step state.
__global__ __launch_bounds__(1024, 1) void k_gru(const bf16_t* __restrict__ Whh,
                                                 const float* __restrict__ xp,
                                                 const float* __restrict__ b_hh,
                                                 const int* __restrict__ seqlen,
                                                 float* __restrict__ hT) {
  __shared__ bf16_t hs[128 * 256];   // 64 KB  bf16 mirror of h
  __shared__ bf16_t hp[128 * 768];   // 192 KB h @ Whhᵀ for current step

  int tid = threadIdx.x, lane = tid & 31, wave = tid >> 5;
  int mt = wave >> 2;                // 0..7 : output rows mt*16..+15
  int ng = wave & 3;                 // 0..3 : N-tiles ng*12..+11

  for (int i = tid; i < 128 * 256; i += 1024) hs[i] = (bf16_t)0.f;

  float hreg[32];                    // fp32 master state, pair idx = tid + j*1024
#pragma unroll
  for (int j = 0; j < 32; ++j) hreg[j] = 0.f;
  __syncthreads();

  for (int t = 0; t < 1024; ++t) {
    // ---- WMMA phase: hp = bf16(h) @ Whhᵀ ----
#pragma unroll
    for (int batch = 0; batch < 2; ++batch) {
      v8f acc[6];
#pragma unroll
      for (int q = 0; q < 6; ++q) acc[q] = vzero8();
#pragma unroll
      for (int kk = 0; kk < 256; kk += 32) {
        v16bf a = frag_bf16(&hs[(size_t)(mt * 16) * 256 + kk], 256, lane);
#pragma unroll
        for (int q = 0; q < 6; ++q) {
          int n0 = (ng * 12 + batch * 6 + q) * 16;
          v16bf b = frag_bf16(Whh + (size_t)n0 * 256 + kk, 256, lane);
          acc[q] = wmma_bf(a, b, acc[q]);
        }
      }
      int nlo = lane & 15, mhi = (lane >> 4) * 8;
#pragma unroll
      for (int q = 0; q < 6; ++q) {
        int col = (ng * 12 + batch * 6 + q) * 16 + nlo;
#pragma unroll
        for (int e = 0; e < 8; ++e)
          hp[(size_t)(mt * 16 + mhi + e) * 768 + col] = (bf16_t)acc[q][e];
      }
    }
    __syncthreads();

    // ---- gate phase (fp32) ----
#pragma unroll
    for (int j = 0; j < 32; ++j) {
      int idx = tid + j * 1024;
      int b = idx >> 8, e = idx & 255;
      size_t xb = ((size_t)b * 1024 + t) * 768;
      float xr = xp[xb + e], xz = xp[xb + 256 + e], xn = xp[xb + 512 + e];
      float hr = (float)hp[(size_t)b * 768 + e]       + b_hh[e];
      float hz = (float)hp[(size_t)b * 768 + 256 + e] + b_hh[256 + e];
      float hn = (float)hp[(size_t)b * 768 + 512 + e] + b_hh[512 + e];
      float r = sigmoidf_(xr + hr);
      float z = sigmoidf_(xz + hz);
      float n = tanhf(xn + r * hn);
      float hnew = (1.f - z) * n + z * hreg[j];
      if (t < seqlen[b]) hreg[j] = hnew;            // freeze finished sequences
      hs[idx] = (bf16_t)hreg[j];
    }
    __syncthreads();
  }

#pragma unroll
  for (int j = 0; j < 32; ++j) hT[tid + j * 1024] = hreg[j];
}

// ------------------- head 1: trans = tanh(hT@w_linᵀ+b); feat ---------------
// Wlin (128x256 bf16 = 64 KB) staged into LDS via async DMA (8x reuse).
__global__ __launch_bounds__(256) void k_head1(const float* __restrict__ hT,
                                               const bf16_t* __restrict__ Wlin,
                                               const float* __restrict__ b_lin,
                                               const float* __restrict__ noise,
                                               bf16_t* __restrict__ feat) {
  __shared__ bf16_t Bs[128 * 256];                 // 64 KB
  int lane = threadIdx.x & 31, wave = threadIdx.x >> 5;
  int r0 = wave * 16;                              // 8 waves cover M=128

  {
    unsigned lb = lds_off(&Bs[0]);
    const char* gsrc = (const char*)Wlin;
#pragma unroll
    for (int i = 0; i < 16; ++i) {                 // 4096 chunks, 16/thread
      int chunk = threadIdx.x + i * 256;
      async_g2l_b128(lb + chunk * 16, gsrc + (size_t)chunk * 16);
    }
    wait_async0();
  }
  __syncthreads();

  v8f acc[8];
#pragma unroll
  for (int j = 0; j < 8; ++j) acc[j] = vzero8();

#pragma unroll
  for (int kk = 0; kk < 256; kk += 32) {
    v16bf a = frag_f32(hT + (size_t)r0 * 256 + kk, 256, lane);
#pragma unroll
    for (int j = 0; j < 8; ++j) {
      v16bf b = frag_bf16(&Bs[(size_t)(j * 16) * 256 + kk], 256, lane);
      acc[j] = wmma_bf(a, b, acc[j]);
    }
  }
  int nlo = lane & 15, mhi = (lane >> 4) * 8;
#pragma unroll
  for (int j = 0; j < 8; ++j) {
    int c = j * 16 + nlo;                          // 0..127 -> trans columns
    float bias = b_lin[c];
#pragma unroll
    for (int e = 0; e < 8; ++e) {
      int r = r0 + mhi + e;
      feat[(size_t)r * 256 + c] = (bf16_t)tanhf(acc[j][e] + bias);
    }
  }
  // concat noise into feat columns 128..255
  for (int i = threadIdx.x; i < 128 * 128; i += 256) {
    int b = i >> 7, e2 = i & 127;
    feat[(size_t)b * 256 + 128 + e2] = (bf16_t)noise[i];
  }
}

// --------------- head 2: y[128,25088] = feat @ W1Tᵀ + b1 (WMMA) ------------
// Each block's B-tile (128x256 bf16 = 64 KB) staged into LDS (8x reuse).
__global__ __launch_bounds__(256) void k_head2(const bf16_t* __restrict__ feat,
                                               const bf16_t* __restrict__ W1T,
                                               const float* __restrict__ b1,
                                               float* __restrict__ y) {
  __shared__ bf16_t Bs[128 * 256];                 // 64 KB
  int lane = threadIdx.x & 31, wave = threadIdx.x >> 5;
  int r0 = wave * 16;                              // 8 waves cover M=128
  int c0 = blockIdx.x * 128;                       // 196 blocks cover N=25088

  {
    unsigned lb = lds_off(&Bs[0]);
    const char* gsrc = (const char*)(W1T + (size_t)c0 * 256);
#pragma unroll
    for (int i = 0; i < 16; ++i) {                 // 4096 chunks, 16/thread
      int chunk = threadIdx.x + i * 256;
      async_g2l_b128(lb + chunk * 16, gsrc + (size_t)chunk * 16);
    }
    wait_async0();
  }
  __syncthreads();

  v8f acc[8];
#pragma unroll
  for (int j = 0; j < 8; ++j) acc[j] = vzero8();

#pragma unroll
  for (int kk = 0; kk < 256; kk += 32) {
    v16bf a = frag_bf16(feat + (size_t)r0 * 256 + kk, 256, lane);
#pragma unroll
    for (int j = 0; j < 8; ++j) {
      v16bf b = frag_bf16(&Bs[(size_t)(j * 16) * 256 + kk], 256, lane);
      acc[j] = wmma_bf(a, b, acc[j]);
    }
  }
  int nlo = lane & 15, mhi = (lane >> 4) * 8;
#pragma unroll
  for (int j = 0; j < 8; ++j) {
    int c = c0 + j * 16 + nlo;
    float bias = b1[c / 49];                       // n = o*49 + h*7 + w
#pragma unroll
    for (int e = 0; e < 8; ++e) {
      int r = r0 + mhi + e;
      y[(size_t)r * 25088 + c] = acc[j][e] + bias;
    }
  }
}

// -------- out: non-overlapping 4x4 ConvTranspose + sigmoid -> [128,28,28] --
__global__ __launch_bounds__(128) void k_out(const float* __restrict__ y,
                                             const float* __restrict__ W2,
                                             const float* __restrict__ b2,
                                             float* __restrict__ out) {
  __shared__ float w2s[512 * 16];                  // 32 KB, [c][p*4+q]
  for (int i = threadIdx.x; i < 512 * 16; i += 128) w2s[i] = W2[i];
  __syncthreads();

  int g = blockIdx.x * 128 + threadIdx.x;          // 49 blocks * 128 = 6272 = 128*49
  int b = g / 49, ij = g % 49;
  float acc[16];
  float bias = b2[0];
#pragma unroll
  for (int pq = 0; pq < 16; ++pq) acc[pq] = bias;

  const float* yb = y + (size_t)b * 25088 + ij;    // y[b][c*49 + ij]
  for (int c = 0; c < 512; ++c) {
    float v = yb[(size_t)c * 49];
#pragma unroll
    for (int pq = 0; pq < 16; ++pq) acc[pq] += v * w2s[c * 16 + pq];
  }
  int ii = ij / 7, jj = ij % 7;
#pragma unroll
  for (int pq = 0; pq < 16; ++pq) {
    int p = pq >> 2, q = pq & 3;
    out[(size_t)b * 784 + (ii * 4 + p) * 28 + (jj * 4 + q)] = sigmoidf_(acc[pq]);
  }
}

// ------------------------------- launcher ----------------------------------

extern "C" void kernel_launch(void* const* d_in, const int* in_sizes, int n_in,
                              void* d_out, int out_size, void* d_ws, size_t ws_size,
                              hipStream_t stream) {
  const float* x      = (const float*)d_in[0];   // [128,1024,256]
  const int*   seqlen = (const int*)  d_in[1];   // [128]
  const float* noise  = (const float*)d_in[2];   // [128,128]
  const float* w_ih   = (const float*)d_in[3];   // [768,256]
  const float* w_hh   = (const float*)d_in[4];   // [768,256]
  const float* b_ih   = (const float*)d_in[5];   // [768]
  const float* b_hh   = (const float*)d_in[6];   // [768]
  const float* w_lin  = (const float*)d_in[7];   // [128,256]
  const float* b_lin  = (const float*)d_in[8];   // [128]
  const float* W1     = (const float*)d_in[9];   // [256,512,7,7]
  const float* b1     = (const float*)d_in[10];  // [512]
  const float* W2     = (const float*)d_in[11];  // [512,1,4,4]
  const float* b2     = (const float*)d_in[12];  // [1]
  float* out = (float*)d_out;                    // [128,1,28,28]

  char* ws = (char*)d_ws;                        // needs ~474 MB
  bf16_t* Wcat = (bf16_t*)(ws + 0);
  bf16_t* Wlin = (bf16_t*)(ws + 786432);
  bf16_t* W1T  = (bf16_t*)(ws + 851968);
  bf16_t* Xbf  = (bf16_t*)(ws + 13697024);
  float*  XP   = (float*) (ws + 80805888);
  float*  HT   = (float*) (ws + 483459072ull);
  bf16_t* FEAT = (bf16_t*)(ws + 483590144ull);
  float*  Y    = (float*) (ws + 483655680ull);

  k_pack_wcat<<<dim3(1536), dim3(256), 0, stream>>>(w_ih, w_hh, Wcat);
  k_pack_wlin<<<dim3(128), dim3(256), 0, stream>>>(w_lin, Wlin);
  k_pack_w1t<<<dim3(25088), dim3(256), 0, stream>>>(W1, W1T);
  k_cvt_x<<<dim3(131072), dim3(256), 0, stream>>>(x, Xbf);

  k_xproj<<<dim3(1024, 8), dim3(256), 0, stream>>>(Xbf, Wcat, b_ih, XP);
  k_gru<<<dim3(1), dim3(1024), 0, stream>>>(Wcat + 768 * 256, XP, b_hh, seqlen, HT);
  k_head1<<<dim3(1), dim3(256), 0, stream>>>(HT, Wlin, b_lin, noise, FEAT);
  k_head2<<<dim3(196), dim3(256), 0, stream>>>(FEAT, W1T, b1, Y);
  k_out<<<dim3(49), dim3(128), 0, stream>>>(Y, W2, b2, out);
}
